// DGCNNBinarySeg_72189810311622
// MI455X (gfx1250) — matrix operand order repaired and checked by reading, not compiled, not run on hardware
//
#include <hip/hip_runtime.h>
#include <hip/hip_bf16.h>
#include <math.h>
#include <stdint.h>

#define Nn   65536
#define Bb   16
#define NPTS 4096
#define KNN  20

typedef __attribute__((ext_vector_type(16))) __bf16 v16bf;
typedef __attribute__((ext_vector_type(8)))  __bf16 v8bf;
typedef __attribute__((ext_vector_type(8)))  float  v8f;
typedef __attribute__((ext_vector_type(4)))  unsigned int u32x4;
typedef __attribute__((ext_vector_type(8)))  int i32x8;
typedef __attribute__((ext_vector_type(4)))  int i32x4;

// ---------------------------------------------------------------- WMMA helpers
__device__ __forceinline__ v8f wmma_bf16(v16bf a, v16bf b, v8f c) {
  // D = A(16x32) x B(32x16) + C, f32 accumulate
  return __builtin_amdgcn_wmma_f32_16x16x32_bf16(false, a, false, b,
                                                 (short)0, c, false, false);
}

__device__ __forceinline__ v16bf load_frag16(const __bf16* p0, const __bf16* p1) {
  v8bf lo = *(const v8bf*)p0;
  v8bf hi = *(const v8bf*)p1;
  return __builtin_shufflevector(lo, hi, 0,1,2,3,4,5,6,7,8,9,10,11,12,13,14,15);
}

// A fragment from a row-major MxK tile (bf16, stride lda halves).
// Layout (matches B): lane<16 -> row=lane, K = 0..15;
// lane>=16 -> row=lane-16, K = 16..31.
__device__ __forceinline__ v16bf load_a_frag(const __bf16* tile, int lda,
                                             int kc, int lane) {
  int row = lane & 15;
  int kb  = (lane >> 4) << 4;
  const __bf16* p = tile + (size_t)row * lda + (kc << 5) + kb;
  return load_frag16(p, p + 8);
}

// B fragment from W^T stored row-major (Nc x Kdim, stride ldb halves).
// Lane holds B column n; lanes<16 hold K 0..15 of the 32-chunk, lanes>=16
// hold K 16..31 -> one contiguous 32B read per lane.
__device__ __forceinline__ v16bf load_b_frag(const __bf16* wt, int ldb,
                                             int n, int kc, int lane) {
  const __bf16* p = wt + (size_t)n * ldb + (kc << 5) + ((lane >> 4) << 4);
  return *(const v16bf*)p;
}

// ---------------------------------------------------------------- TDM staging
// One-shot 2D tensor DMA: whole (rows x cols) bf16 matrix, global -> LDS.
// D# per CDNA5 ISA §8.3/8.4: count=1, type=2, data_size=2B,
// tensor_dim0 = tile_dim0 = cols, tensor_dim1 = tile_dim1 = rows,
// tensor_dim0_stride = cols; remaining groups zero (2D case).
__device__ __forceinline__ void tdm_load_2d(unsigned lds_off, const void* gaddr,
                                            unsigned rows, unsigned cols) {
  unsigned long long ga = (unsigned long long)(uintptr_t)gaddr;
  unsigned glo = (unsigned)ga, ghi = (unsigned)(ga >> 32);
  u32x4 g0;
  g0[0] = 1u;                                   // count=1, user descriptor
  g0[1] = lds_off;                              // lds_addr (bytes)
  g0[2] = glo;                                  // global_addr[31:0]
  g0[3] = (ghi & 0x01FFFFFFu) | (2u << 30);     // global_addr[56:32] | type=2
  i32x8 g1;
  g1[0] = (int)(1u << 16);                      // data_size = 2 bytes
  g1[1] = (int)((cols & 0xFFFFu) << 16);        // tensor_dim0[15:0]
  g1[2] = (int)((cols >> 16) | ((rows & 0xFFFFu) << 16));  // dim0 hi | dim1 lo
  g1[3] = (int)((rows >> 16) | ((cols & 0xFFFFu) << 16));  // dim1 hi | tile_dim0
  g1[4] = (int)(rows & 0xFFFFu);                // tile_dim1 (tile_dim2 = 0)
  g1[5] = (int)cols;                            // tensor_dim0_stride[31:0]
  g1[6] = 0;                                    // stride hi / dim1_stride lo
  g1[7] = 0;
  i32x4 z4 = {0, 0, 0, 0};
  i32x8 z8 = {0, 0, 0, 0, 0, 0, 0, 0};
  // 6-arg form (clang-23 / therock-10.0 headers)
  __builtin_amdgcn_tensor_load_to_lds(g0, g1, z4, z4, z8, 0);
}

// ---------------------------------------------------------------- weight prep
// w: (Kdim x Nc) f32 row-major  ->  wt: (Nc x Kdim) bf16 row-major (= W^T)
__global__ void convert_wt_kernel(const float* __restrict__ w,
                                  __bf16* __restrict__ wt, int Kdim, int Nc) {
  int i = blockIdx.x * blockDim.x + threadIdx.x;
  if (i >= Kdim * Nc) return;
  int n = i / Kdim, k = i % Kdim;
  wt[i] = (__bf16)w[(size_t)k * Nc + n];
}

// ---------------------------------------------------------------- x0 = in @ W0 + b0
__global__ void input_proj_kernel(const float* __restrict__ xyz,
                                  const float* __restrict__ xf,
                                  const float* __restrict__ W0,
                                  const float* __restrict__ b0,
                                  __bf16* __restrict__ x0b) {
  int i = blockIdx.x * blockDim.x + threadIdx.x;   // Nn*64 threads
  int n = i >> 6, c = i & 63;
  float in[10];
  in[0] = xyz[n*3+0]; in[1] = xyz[n*3+1]; in[2] = xyz[n*3+2];
  #pragma unroll
  for (int t = 0; t < 7; ++t) in[3+t] = xf[n*7+t];
  float s = b0[c];
  #pragma unroll
  for (int t = 0; t < 10; ++t) s += in[t] * W0[t*64 + c];
  x0b[(size_t)n*64 + c] = (__bf16)s;
}

// ---------------------------------------------------------------- KNN (top-20)
__launch_bounds__(256)
__global__ void knn_kernel(const float* __restrict__ pos, int* __restrict__ nbr) {
  __shared__ float cp[256*3];
  int g  = blockIdx.x >> 4;        // graph
  int qc = blockIdx.x & 15;        // query chunk
  int q  = qc*256 + threadIdx.x;   // local query id
  size_t gq = (size_t)g*NPTS + q;
  float qx = pos[gq*3+0], qy = pos[gq*3+1], qz = pos[gq*3+2];
  float bd[KNN]; int bi[KNN];
  #pragma unroll
  for (int j = 0; j < KNN; ++j) { bd[j] = 1e30f; bi[j] = 0; }
  for (int ct = 0; ct < 16; ++ct) {
    __syncthreads();
    for (int t = threadIdx.x; t < 256*3; t += 256)
      cp[t] = pos[((size_t)g*NPTS + ct*256)*3 + t];
    __syncthreads();
    for (int i = 0; i < 256; ++i) {
      int ci = ct*256 + i;
      float dx = qx - cp[i*3+0], dy = qy - cp[i*3+1], dz = qz - cp[i*3+2];
      float d = dx*dx + dy*dy + dz*dz;
      if (ci == q) continue;                 // exclude self (diag +1e9 in ref)
      if (d < bd[KNN-1]) {                   // register insertion (no scratch)
        int cidx = g*NPTS + ci;
        #pragma unroll
        for (int j = 0; j < KNN; ++j) {
          if (d < bd[j]) {
            float td = bd[j]; bd[j] = d; d = td;
            int   ti = bi[j]; bi[j] = cidx; cidx = ti;
          }
        }
      }
    }
  }
  #pragma unroll
  for (int j = 0; j < KNN; ++j) nbr[gq*KNN + j] = bi[j];
}

// ---------------------------------------------------------------- EdgeConv
// 16 points/block, 320 edges = 20 M-tiles; fused gather+MLP(2)+max-pool.
// LDSW: stage both weight matrices in LDS via the Tensor Data Mover (20x reuse
// per block); otherwise stream B from L2 with software pipelining.
template<int CIN, int CMID, int COUT, bool LDSW>
__launch_bounds__(128)
__global__ void edge_conv_kernel(const __bf16* __restrict__ src, int ld,
                                 const int* __restrict__ nbr,
                                 const __bf16* __restrict__ W1t,
                                 const float* __restrict__ b1,
                                 const __bf16* __restrict__ W2t,
                                 const float* __restrict__ b2,
                                 float* __restrict__ mout) {
  constexpr int FD  = 2*CIN;
  constexpr int NT1 = CMID/16, NT2 = COUT/16;
  constexpr int KC1 = FD/32,   KC2 = CMID/32;
  __shared__ __bf16 xi[16*CIN];
  __shared__ __bf16 feat[16*FD];
  __shared__ __bf16 m1[16*CMID];
  __shared__ float  acc[16*COUT];
  __shared__ __bf16 w1s[LDSW ? CMID*FD   : 1];
  __shared__ __bf16 w2s[LDSW ? COUT*CMID : 1];
  const int tid  = threadIdx.x;
  const int lane = tid & 31, wave = tid >> 5;
  const int p0   = blockIdx.x * 16;

  if constexpr (LDSW) {
    if (wave == 0) {                 // TDM is per-wave, EXEC-independent
      tdm_load_2d((unsigned)(uintptr_t)(void*)w1s, W1t, CMID, FD);
      tdm_load_2d((unsigned)(uintptr_t)(void*)w2s, W2t, COUT, CMID);
      __builtin_amdgcn_s_wait_tensorcnt(0);
    }
  } else {
    // keep weights hot in L2 (global_prefetch_b8)
    for (size_t i = (size_t)tid*64; i < (size_t)CMID*FD;   i += 128*64)
      __builtin_prefetch(W1t + i, 0, 1);
    for (size_t i = (size_t)tid*64; i < (size_t)COUT*CMID; i += 128*64)
      __builtin_prefetch(W2t + i, 0, 1);
  }
  const __bf16* W1p = LDSW ? (const __bf16*)w1s : W1t;
  const __bf16* W2p = LDSW ? (const __bf16*)w2s : W2t;

  for (int t = tid; t < 16*CIN; t += 128) {
    int p = t / CIN, c = t % CIN;
    xi[t] = src[(size_t)(p0+p)*ld + c];
  }
  for (int t = tid; t < 16*COUT; t += 128) acc[t] = 0.f;  // post-relu max >= 0
  __syncthreads();                 // also publishes TDM-loaded weights

  for (int mt = 0; mt < 20; ++mt) {
    // build feat = [x_i | x_j - x_i] for 16 edges
    for (int t = tid; t < 16*FD; t += 128) {
      int m = t / FD, c = t % FD;
      int e = mt*16 + m;
      int p = e / KNN, k = e % KNN;
      int j = nbr[(size_t)(p0+p)*KNN + k];
      int cc = (c < CIN) ? c : (c - CIN);
      float xiv = (float)xi[p*CIN + cc];
      float v = (c < CIN) ? xiv
                          : ((float)src[(size_t)j*ld + cc] - xiv);
      feat[m*FD + c] = (__bf16)v;
    }
    __syncthreads();
    // layer 1: feat(16xFD) @ W1 -> relu -> m1 (LDS, bf16)
    for (int nt = wave; nt < NT1; nt += 4) {
      int n = nt*16 + (lane & 15);
      float bv = b1[n];
      v8f c;
      #pragma unroll
      for (int r = 0; r < 8; ++r) c[r] = bv;
      v16bf b = load_b_frag(W1p, FD, n, 0, lane);
      #pragma unroll
      for (int kc = 0; kc < KC1; ++kc) {
        v16bf bn = (kc + 1 < KC1) ? load_b_frag(W1p, FD, n, kc+1, lane) : b;
        v16bf a  = load_a_frag(feat, FD, kc, lane);
        c = wmma_bf16(a, b, c);
        b = bn;
      }
      int mrow = (lane >> 4) << 3;
      #pragma unroll
      for (int r = 0; r < 8; ++r)
        m1[(mrow+r)*CMID + n] = (__bf16)fmaxf(c[r], 0.f);
    }
    __syncthreads();
    // layer 2: m1 @ W2 -> relu -> max-pool per point into acc
    for (int nt = wave; nt < NT2; nt += 4) {
      int n = nt*16 + (lane & 15);
      float bv = b2[n];
      v8f c;
      #pragma unroll
      for (int r = 0; r < 8; ++r) c[r] = bv;
      v16bf b = load_b_frag(W2p, CMID, n, 0, lane);
      #pragma unroll
      for (int kc = 0; kc < KC2; ++kc) {
        v16bf bn = (kc + 1 < KC2) ? load_b_frag(W2p, CMID, n, kc+1, lane) : b;
        v16bf a  = load_a_frag(m1, CMID, kc, lane);
        c = wmma_bf16(a, b, c);
        b = bn;
      }
      // two-phase: lanes l and l+16 share column n but rows may hit the
      // same point; divergent halves serialize within the wave -> race-free.
      if (lane < 16) {
        #pragma unroll
        for (int r = 0; r < 8; ++r) {
          int p = (mt*16 + r) / KNN;
          float* ap = &acc[p*COUT + n];
          *ap = fmaxf(*ap, fmaxf(c[r], 0.f));
        }
      }
      if (lane >= 16) {
        #pragma unroll
        for (int r = 0; r < 8; ++r) {
          int p = (mt*16 + 8 + r) / KNN;
          float* ap = &acc[p*COUT + n];
          *ap = fmaxf(*ap, fmaxf(c[r], 0.f));
        }
      }
    }
    __syncthreads();
  }
  for (int t = tid; t < 16*COUT; t += 128) {
    int p = t / COUT, c = t % COUT;
    mout[(size_t)(p0+p)*COUT + c] = acc[t];
  }
}

// ---------------------------------------------------------------- BatchNorm
__global__ void bn_stats_kernel(const float* __restrict__ m, int C,
                                float* __restrict__ stats) {
  int c = threadIdx.x;                     // blockDim.x == C
  size_t r0 = (size_t)blockIdx.x * 256;
  float s = 0.f, ss = 0.f;
  for (int r = 0; r < 256; ++r) {
    float v = m[(r0+r)*(size_t)C + c];
    s += v; ss += v*v;
  }
  atomicAdd(&stats[c], s);
  atomicAdd(&stats[C+c], ss);
}

__global__ void bn_apply_kernel(const float* __restrict__ m, int C,
                                const float* __restrict__ stats,
                                const float* __restrict__ g,
                                const float* __restrict__ bt,
                                __bf16* __restrict__ dst, int ldd) {
  size_t i = (size_t)blockIdx.x * blockDim.x + threadIdx.x;
  if (i >= (size_t)Nn * C) return;
  size_t n = i / C; int c = (int)(i % C);
  float mu  = stats[c] * (1.f/Nn);
  float var = stats[C+c] * (1.f/Nn) - mu*mu;
  float v = (m[i] - mu) * rsqrtf(var + 1e-5f) * g[c] + bt[c];
  dst[n*(size_t)ldd + c] = (__bf16)v;
}

// ---------------------------------------------------------------- h_local MLP
__launch_bounds__(128)
__global__ void local_mlp_kernel(const __bf16* __restrict__ h,      // N x 512
                                 const __bf16* __restrict__ W1t,
                                 const float*  __restrict__ b1,
                                 const __bf16* __restrict__ W2t,
                                 const float*  __restrict__ b2,
                                 __bf16* __restrict__ out) {        // N x 256
  __shared__ __bf16 t1[16*256];
  const int tid = threadIdx.x, lane = tid & 31, wave = tid >> 5;
  const size_t r0 = (size_t)blockIdx.x * 16;
  const __bf16* A = h + r0 * 512;
  for (int nt = wave; nt < 16; nt += 4) {
    int n = nt*16 + (lane & 15);
    float bv = b1[n];
    v8f c;
    #pragma unroll
    for (int r = 0; r < 8; ++r) c[r] = bv;
    v16bf b = load_b_frag(W1t, 512, n, 0, lane);
    #pragma unroll
    for (int kc = 0; kc < 16; ++kc) {
      v16bf bn = (kc + 1 < 16) ? load_b_frag(W1t, 512, n, kc+1, lane) : b;
      v16bf a  = load_a_frag(A, 512, kc, lane);
      c = wmma_bf16(a, b, c);
      b = bn;
    }
    int mrow = (lane >> 4) << 3;
    #pragma unroll
    for (int r = 0; r < 8; ++r)
      t1[(mrow+r)*256 + n] = (__bf16)fmaxf(c[r], 0.f);
  }
  __syncthreads();
  for (int nt = wave; nt < 16; nt += 4) {
    int n = nt*16 + (lane & 15);
    float bv = b2[n];
    v8f c;
    #pragma unroll
    for (int r = 0; r < 8; ++r) c[r] = bv;
    v16bf b = load_b_frag(W2t, 256, n, 0, lane);
    #pragma unroll
    for (int kc = 0; kc < 8; ++kc) {
      v16bf bn = (kc + 1 < 8) ? load_b_frag(W2t, 256, n, kc+1, lane) : b;
      v16bf a  = load_a_frag(t1, 256, kc, lane);
      c = wmma_bf16(a, b, c);
      b = bn;
    }
    int mrow = (lane >> 4) << 3;
    #pragma unroll
    for (int r = 0; r < 8; ++r)
      out[(r0 + mrow + r)*256 + n] = (__bf16)fmaxf(c[r], 0.f);
  }
}

// ---------------------------------------------------------------- global paths
__global__ void graph_max_kernel(const __bf16* __restrict__ hl,
                                 float* __restrict__ g0) {
  int b = blockIdx.x, c = threadIdx.x;     // 16 x 256
  float v = 0.f;                           // h_local is post-relu
  for (int r = 0; r < NPTS; ++r)
    v = fmaxf(v, (float)hl[((size_t)b*NPTS + r)*256 + c]);
  g0[b*256 + c] = v;
}

__global__ void gproj_kernel(const float* __restrict__ g0,
                             const float* __restrict__ W,
                             const float* __restrict__ bias,
                             __bf16* __restrict__ out) {
  int b = blockIdx.x, c = threadIdx.x;
  float s = bias[c];
  for (int i = 0; i < 256; ++i) s += g0[b*256+i] * W[i*256 + c];
  out[b*256 + c] = (__bf16)fmaxf(s, 0.f);
}

__global__ void pretrain_kernel(const float* __restrict__ pg,   // B x 1280
                                const float* __restrict__ pW,
                                const float* __restrict__ pb,
                                const float* __restrict__ g,
                                const float* __restrict__ bt,
                                __bf16* __restrict__ out) {     // B x 256 (SiLU)
  __shared__ float red[256];
  int b = blockIdx.x, c = threadIdx.x;
  float s = pb[c];
  for (int i = 0; i < 1280; ++i) s += pg[b*1280+i] * pW[(size_t)i*256 + c];
  red[c] = s; __syncthreads();
  for (int st = 128; st > 0; st >>= 1) {
    if (c < st) red[c] += red[c+st];
    __syncthreads();
  }
  float mu = red[0] * (1.f/256.f); __syncthreads();
  float dv = s - mu;
  red[c] = dv*dv; __syncthreads();
  for (int st = 128; st > 0; st >>= 1) {
    if (c < st) red[c] += red[c+st];
    __syncthreads();
  }
  float var = red[0] * (1.f/256.f);
  float y = dv * rsqrtf(var + 1e-5f) * g[c] + bt[c];
  y = y / (1.f + expf(-y));                // SiLU
  out[b*256 + c] = (__bf16)y;
}

// ---------------------------------------------------------------- head
__launch_bounds__(128)
__global__ void head_kernel(const __bf16* __restrict__ hl,   // N x 256
                            const __bf16* __restrict__ gg,   // B x 256
                            const __bf16* __restrict__ gp,   // B x 256
                            const __bf16* __restrict__ W1t,  // 256 x 768
                            const float*  __restrict__ b1,
                            const float*  __restrict__ w2,   // 256
                            const float*  __restrict__ b2,   // 1
                            float* __restrict__ out) {
  __shared__ __bf16 A[16*768];
  __shared__ float  T[16*256];
  __shared__ float  part[16*8];
  const int tid = threadIdx.x, lane = tid & 31, wave = tid >> 5;
  const size_t r0 = (size_t)blockIdx.x * 16;
  const int b = (int)(r0 / NPTS);          // tiles never straddle graphs
  for (int t = tid; t < 16*768; t += 128) {
    int m = t / 768, c = t % 768;
    __bf16 v;
    if (c < 256)      v = hl[(r0+m)*256 + c];
    else if (c < 512) v = gg[b*256 + (c-256)];
    else              v = gp[b*256 + (c-512)];
    A[t] = v;
  }
  __syncthreads();
  for (int nt = wave; nt < 16; nt += 4) {
    int n = nt*16 + (lane & 15);
    float bv = b1[n];
    v8f c;
    #pragma unroll
    for (int r = 0; r < 8; ++r) c[r] = bv;
    v16bf b = load_b_frag(W1t, 768, n, 0, lane);
    #pragma unroll
    for (int kc = 0; kc < 24; ++kc) {
      v16bf bn = (kc + 1 < 24) ? load_b_frag(W1t, 768, n, kc+1, lane) : b;
      v16bf a  = load_a_frag(A, 768, kc, lane);
      c = wmma_bf16(a, b, c);
      b = bn;
    }
    int mrow = (lane >> 4) << 3;
    #pragma unroll
    for (int r = 0; r < 8; ++r)
      T[(mrow+r)*256 + n] = fmaxf(c[r], 0.f);
  }
  __syncthreads();
  {
    int p = tid >> 3, ch = tid & 7;
    float s = 0.f;
    for (int i = 0; i < 32; ++i) { int c = ch*32 + i; s += T[p*256+c] * w2[c]; }
    part[p*8 + ch] = s;
  }
  __syncthreads();
  if (tid < 16) {
    float s = b2[0];
    #pragma unroll
    for (int ch = 0; ch < 8; ++ch) s += part[tid*8 + ch];
    out[r0 + tid] = s;
  }
}

// ---------------------------------------------------------------- launch
extern "C" void kernel_launch(void* const* d_in, const int* in_sizes, int n_in,
                              void* d_out, int out_size, void* d_ws, size_t ws_size,
                              hipStream_t stream) {
  (void)in_sizes; (void)n_in; (void)out_size; (void)ws_size;
  const float* xyz = (const float*)d_in[0];
  const float* xf  = (const float*)d_in[1];
  const float* pos = (const float*)d_in[2];
  const float* pg  = (const float*)d_in[4];
  const float* W0  = (const float*)d_in[5],  *b0   = (const float*)d_in[6];
  const float* e1W1= (const float*)d_in[7],  *e1b1 = (const float*)d_in[8];
  const float* e1W2= (const float*)d_in[9],  *e1b2 = (const float*)d_in[10];
  const float* e1g = (const float*)d_in[11], *e1bt = (const float*)d_in[12];
  const float* e2W1= (const float*)d_in[13], *e2b1 = (const float*)d_in[14];
  const float* e2W2= (const float*)d_in[15], *e2b2 = (const float*)d_in[16];
  const float* e2g = (const float*)d_in[17], *e2bt = (const float*)d_in[18];
  const float* e3W1= (const float*)d_in[19], *e3b1 = (const float*)d_in[20];
  const float* e3W2= (const float*)d_in[21], *e3b2 = (const float*)d_in[22];
  const float* e3g = (const float*)d_in[23], *e3bt = (const float*)d_in[24];
  const float* l1W = (const float*)d_in[25], *l1b  = (const float*)d_in[26];
  const float* l2W = (const float*)d_in[27], *l2b  = (const float*)d_in[28];
  const float* gW  = (const float*)d_in[29], *gb   = (const float*)d_in[30];
  const float* pW  = (const float*)d_in[31], *pb   = (const float*)d_in[32];
  const float* plng= (const float*)d_in[33], *plnb = (const float*)d_in[34];
  const float* h1W = (const float*)d_in[35], *h1b  = (const float*)d_in[36];
  const float* h2W = (const float*)d_in[37], *h2b  = (const float*)d_in[38];
  float* out = (float*)d_out;

  char* ws = (char*)d_ws;
  size_t off = 0;
  auto carve = [&](size_t bytes) {
    size_t o = off; off += (bytes + 255) & ~(size_t)255; return o;
  };
  int*    nbr    = (int*)   (ws + carve((size_t)Nn*KNN*4));
  __bf16* x0b    = (__bf16*)(ws + carve((size_t)Nn*64*2));
  __bf16* hcatb  = (__bf16*)(ws + carve((size_t)Nn*512*2));   // h1|h2|h3 bf16
  float*  mbuf   = (float*) (ws + carve((size_t)Nn*256*4));   // pre-BN, reused
  __bf16* hlb    = (__bf16*)(ws + carve((size_t)Nn*256*2));   // h_local bf16
  float*  stats  = (float*) (ws + carve(512*4));
  float*  ggeom0 = (float*) (ws + carve(Bb*256*4));
  __bf16* ggeomb = (__bf16*)(ws + carve(Bb*256*2));
  __bf16* gsb    = (__bf16*)(ws + carve(Bb*256*2));
  __bf16* e1W1t  = (__bf16*)(ws + carve(96*128*2));
  __bf16* e1W2t  = (__bf16*)(ws + carve(96*96*2));
  __bf16* e2W1t  = (__bf16*)(ws + carve(160*192*2));
  __bf16* e2W2t  = (__bf16*)(ws + carve(160*160*2));
  __bf16* e3W1t  = (__bf16*)(ws + carve(256*320*2));
  __bf16* e3W2t  = (__bf16*)(ws + carve(256*256*2));
  __bf16* l1Wt   = (__bf16*)(ws + carve(256*512*2));
  __bf16* l2Wt   = (__bf16*)(ws + carve(256*256*2));
  __bf16* h1Wt   = (__bf16*)(ws + carve(256*768*2));

  auto cvt = [&](const float* w, __bf16* wt, int K, int Ncol) {
    int tot = K * Ncol;
    convert_wt_kernel<<<(tot+255)/256, 256, 0, stream>>>(w, wt, K, Ncol);
  };
  cvt(e1W1, e1W1t, 128, 96);  cvt(e1W2, e1W2t, 96, 96);
  cvt(e2W1, e2W1t, 192, 160); cvt(e2W2, e2W2t, 160, 160);
  cvt(e3W1, e3W1t, 320, 256); cvt(e3W2, e3W2t, 256, 256);
  cvt(l1W,  l1Wt,  512, 256); cvt(l2W,  l2Wt,  256, 256);
  cvt(h1W,  h1Wt,  768, 256);

  input_proj_kernel<<<(Nn*64)/256, 256, 0, stream>>>(xyz, xf, W0, b0, x0b);
  knn_kernel<<<Bb*16, 256, 0, stream>>>(pos, nbr);

  // EdgeConv 1: x0(64) -> 96   (h1 at hcatb col 0), TDM-staged weights
  edge_conv_kernel<64,96,96,true><<<Nn/16, 128, 0, stream>>>(
      x0b, 64, nbr, e1W1t, e1b1, e1W2t, e1b2, mbuf);
  (void)hipMemsetAsync(stats, 0, 2*96*4, stream);
  bn_stats_kernel<<<Nn/256, 96, 0, stream>>>(mbuf, 96, stats);
  bn_apply_kernel<<<(int)(((size_t)Nn*96+255)/256), 256, 0, stream>>>(
      mbuf, 96, stats, e1g, e1bt, hcatb + 0, 512);

  // EdgeConv 2: h1(96) -> 160  (h2 at hcatb col 96), TDM-staged weights
  edge_conv_kernel<96,160,160,true><<<Nn/16, 128, 0, stream>>>(
      hcatb + 0, 512, nbr, e2W1t, e2b1, e2W2t, e2b2, mbuf);
  (void)hipMemsetAsync(stats, 0, 2*160*4, stream);
  bn_stats_kernel<<<Nn/256, 160, 0, stream>>>(mbuf, 160, stats);
  bn_apply_kernel<<<(int)(((size_t)Nn*160+255)/256), 256, 0, stream>>>(
      mbuf, 160, stats, e2g, e2bt, hcatb + 96, 512);

  // EdgeConv 3: h2(160) -> 256 (h3 at hcatb col 256), weights from L2 (too
  // large for LDS alongside working set)
  edge_conv_kernel<160,256,256,false><<<Nn/16, 128, 0, stream>>>(
      hcatb + 96, 512, nbr, e3W1t, e3b1, e3W2t, e3b2, mbuf);
  (void)hipMemsetAsync(stats, 0, 2*256*4, stream);
  bn_stats_kernel<<<Nn/256, 256, 0, stream>>>(mbuf, 256, stats);
  bn_apply_kernel<<<(int)(((size_t)Nn*256+255)/256), 256, 0, stream>>>(
      mbuf, 256, stats, e3g, e3bt, hcatb + 256, 512);

  // h_local = relu(relu(h @ l1W) @ l2W)
  local_mlp_kernel<<<Nn/16, 128, 0, stream>>>(hcatb, l1Wt, l1b, l2Wt, l2b, hlb);

  // g_geom = relu(max_pool(h_local) @ gW + gb)
  graph_max_kernel<<<Bb, 256, 0, stream>>>(hlb, ggeom0);
  gproj_kernel<<<Bb, 256, 0, stream>>>(ggeom0, gW, gb, ggeomb);

  // pretrain branch: LN + SiLU
  pretrain_kernel<<<Bb, 256, 0, stream>>>(pg, pW, pb, plng, plnb, gsb);

  // head: relu([h_local|g_geom|gs] @ h1W + b) @ h2W + b
  head_kernel<<<Nn/16, 128, 0, stream>>>(hlb, ggeomb, gsb, h1Wt, h1b,
                                         h2W, h2b, out);
}
